// MemNet_70428873720088
// MI455X (gfx1250) — matrix operand order repaired and checked by reading, not compiled, not run on hardware
//
#include <hip/hip_runtime.h>
#include <hip/hip_bf16.h>

// ---------------------------------------------------------------------------
// MemNet recurrence for MI455X (gfx1250, wave32, WMMA).
// One persistent kernel; 2 grid barriers per timestep.
//  Region A (all WGs): batched bf16 WMMA GEMMs  gi, gh, logits(t-1), M=32.
//  Region B (WG b<32): per-batch GRU gates, key GEMV (WMMA), cosine
//  addressing (WMMA vs bf16 memory mirror), softmax+top-8 (wave32 shfl
//  reductions), sparse erase/add memory write with incremental norm upkeep.
// ---------------------------------------------------------------------------

typedef __bf16 v16bf __attribute__((ext_vector_type(16)));
typedef __bf16 v8bf  __attribute__((ext_vector_type(8)));
typedef float  v8f   __attribute__((ext_vector_type(8)));

union ABf { v16bf v; v8bf h[2]; };

constexpr int kB = 32, kT = 256, kV = 8192, kE = 256, kH = 512, kD = 64,
              kS = 1024, kNH = 4, kTOPK = 8;
constexpr int kX   = kE + kD;       // 320  (GRU input = [emb | read])
constexpr int kG3  = 3 * kH;        // 1536 (gate width)
constexpr int kNK  = 3 * kNH * kD;  // 768  (rk|wk|wv packed)
constexpr float kEPS = 1e-8f;

constexpr int NWG    = 64;          // persistent workgroups (co-resident)
constexpr int NTHR   = 256;         // 8 waves of 32
constexpr int WPW    = NTHR / 32;
constexpr int TWAVES = NWG * WPW;
constexpr int CAP    = 64;          // max kept entries per head (top-8 + ties)

// float -> bf16, round-to-nearest-even
__device__ __forceinline__ __bf16 f2bf(float x) {
  unsigned u = __float_as_uint(x);
  unsigned r = (u + 0x7FFFu + ((u >> 16) & 1u)) >> 16;
  unsigned short s9 = (unsigned short)r;
  return __builtin_bit_cast(__bf16, s9);
}

__device__ __forceinline__ float wredSum(float v) {
#pragma unroll
  for (int o = 16; o > 0; o >>= 1) v += __shfl_xor(v, o, 32);
  return v;
}
__device__ __forceinline__ float wredMax(float v) {
#pragma unroll
  for (int o = 16; o > 0; o >>= 1) v = fmaxf(v, __shfl_xor(v, o, 32));
  return v;
}
__device__ __forceinline__ float sigmoidf(float x) { return 1.f / (1.f + __expf(-x)); }

// Grid barrier (release + arrive + spin + acquire). bar[0]=count bar[1]=gen.
__device__ __forceinline__ void gsync(unsigned* bar) {
  __threadfence();                 // release my stores to device scope
  __syncthreads();
  if (threadIdx.x == 0) {
    volatile unsigned* gen = bar + 1;
    unsigned g = *gen;
    unsigned a = atomicAdd(bar, 1u);
    if (a == (unsigned)(NWG - 1)) {
      *(volatile unsigned*)bar = 0u;
      __threadfence();
      atomicAdd((unsigned*)bar + 1, 1u);
    } else {
      while (*gen == g) { __builtin_amdgcn_s_sleep(2); }
    }
  }
  __syncthreads();
  __threadfence();                 // acquire: invalidate stale L0 lines
}

// Batched GEMM  out[m][n] = sum_k A[m][k]*W[n][k] + bias[n],  M = 32 rows.
// A row-major bf16 [32][lda], W row-major bf16 [N][ldw], tiles striped over
// all waves of the grid. v_wmma_f32_16x16x32_bf16, f32 accumulate.
__device__ __forceinline__ void gemm32(
    const __bf16* __restrict__ A, int lda,
    const __bf16* __restrict__ W, int ldw,
    const float* __restrict__ bias,
    float* __restrict__ out, size_t orow,
    int N, int K, int gwave) {
  const int lane = threadIdx.x & 31;
  const bool hi  = lane >= 16;
  const int lrow = lane & 15;
  const int tiles = (N >> 4) << 1;  // 2 M-tiles per N-tile
  for (int tl = gwave; tl < tiles; tl += TWAVES) {
    const int mt = tl & 1, nt = tl >> 1;
    const __bf16* arow = A + (size_t)(mt * 16 + lrow) * lda + (hi ? 16 : 0);
    const __bf16* wrow = W + (size_t)(nt * 16 + lrow) * ldw + (hi ? 16 : 0);
    v8f acc = {};
    for (int kc = 0; kc < K; kc += 32) {
      ABf a, b;
      a.v    = *(const v16bf*)(arow + kc);      // A: contiguous 16 halfs of A row
      b.v    = *(const v16bf*)(wrow + kc);      // B: contiguous 16 halfs of W row
      acc = __builtin_amdgcn_wmma_f32_16x16x32_bf16(false, a.v, false, b.v,
                                                    (short)0, acc, false, false);
    }
    const int nidx  = nt * 16 + lrow;
    const float bv  = bias[nidx];
    const int mbase = mt * 16 + (hi ? 8 : 0);
    float* op = out + (size_t)mbase * orow + nidx;
#pragma unroll
    for (int r = 0; r < 8; ++r) op[(size_t)r * orow] = acc[r] + bv;
  }
}

// Sparse content addressing for one batch element.
// key: LDS [kNH*kD] f32. Produces per-head kept lists (<=CAP), kept-sum.
__device__ void do_address(
    const float* __restrict__ key, const float* __restrict__ betaRaw,
    const __bf16* __restrict__ memB, const float* __restrict__ invmB,
    float (*lsw)[kS], __bf16 (*lskn)[kD], float* scaleT,
    int* cnt, int (*sl)[CAP], float (*wl)[CAP], float* rs) {
  const int tid  = threadIdx.x;
  const int lane = tid & 31, wv2 = tid >> 5;
  const bool hi  = lane >= 16;
  const int lrow = lane & 15;

  // per-head scale = softplus(beta) / (||key|| + eps)   (wave per head)
  if (wv2 < kNH) {
    float p = 0.f;
    for (int d2 = lane; d2 < kD; d2 += 32) { float x = key[wv2 * kD + d2]; p += x * x; }
    p = wredSum(p);
    if (lane == 0) {
      float bx = betaRaw[wv2];
      float sp = (bx > 20.f) ? bx : log1pf(__expf(bx));
      scaleT[wv2] = sp / (sqrtf(p) + kEPS);
    }
  }
  __syncthreads();
  // Build bf16 A-tile: rows 0..3 = scaled keys, rows 4..15 = 0
  for (int i = tid; i < 16 * kD; i += NTHR) {
    int h2 = i >> 6, d2 = i & 63;
    float x = (h2 < kNH) ? key[h2 * kD + d2] * scaleT[h2] : 0.f;
    lskn[h2][d2] = f2bf(x);
  }
  __syncthreads();
  // sim[h][s] = scaledKey_h . mem_s * invnorm_s   via WMMA (M=heads, N=slots)
  for (int nt = wv2; nt < kS / 16; nt += WPW) {
    const __bf16* arow = &lskn[lrow][hi ? 16 : 0];
    const __bf16* brow = memB + (size_t)(nt * 16 + lrow) * kD + (hi ? 16 : 0);
    v8f acc = {};
#pragma unroll
    for (int kc = 0; kc < kD; kc += 32) {
      ABf a, b;
      a.v    = *(const v16bf*)(arow + kc);
      b.v    = *(const v16bf*)(brow + kc);
      acc = __builtin_amdgcn_wmma_f32_16x16x32_bf16(false, a.v, false, b.v,
                                                    (short)0, acc, false, false);
    }
    if (lane < 16) {
      int s2 = nt * 16 + lane;
      float im = invmB[s2];
      lsw[0][s2] = acc[0] * im; lsw[1][s2] = acc[1] * im;
      lsw[2][s2] = acc[2] * im; lsw[3][s2] = acc[3] * im;
    }
  }
  __syncthreads();
  // softmax + top-8 threshold + compaction: one wave32 per head, shfl only
  if (wv2 < kNH) {
    float* wr = lsw[wv2];
    float mx = -3.0e38f;
    for (int s2 = lane; s2 < kS; s2 += 32) mx = fmaxf(mx, wr[s2]);
    mx = wredMax(mx);
    float sm = 0.f;
    for (int s2 = lane; s2 < kS; s2 += 32) { float e = __expf(wr[s2] - mx); wr[s2] = e; sm += e; }
    sm = wredSum(sm);
    float inv = 1.f / sm;
    for (int s2 = lane; s2 < kS; s2 += 32) wr[s2] *= inv;
    float cur = 3.0e38f;                       // 8th-largest by max extraction
    for (int it = 0; it < kTOPK; ++it) {
      float m2 = -3.0e38f;
      for (int s2 = lane; s2 < kS; s2 += 32) { float x = wr[s2]; if (x < cur) m2 = fmaxf(m2, x); }
      cur = wredMax(m2);
    }
    if (lane == 0) cnt[wv2] = 0;
    float ks = 0.f;
    for (int s2 = lane; s2 < kS; s2 += 32) {
      float x = wr[s2];
      if (x >= cur) {
        int ix = atomicAdd(&cnt[wv2], 1);
        if (ix < CAP) { sl[wv2][ix] = s2; wl[wv2][ix] = x; }
        ks += x;
      }
    }
    ks = wredSum(ks);
    if (lane == 0) rs[wv2] = ks;
  }
  __syncthreads();
}

// ---------------------------------------------------------------------------
// Prep: bf16 weight conversion, per-batch memory replication + norms,
// zero state, xcat for t=0, barrier reset. Re-run every call (determinism).
// ---------------------------------------------------------------------------
__global__ __launch_bounds__(256) void prep_kernel(
    const int* __restrict__ seq, const float* __restrict__ embed,
    const float* __restrict__ W_ih, const float* __restrict__ W_hh,
    const float* __restrict__ W_rk, const float* __restrict__ b_rk,
    const float* __restrict__ W_wk, const float* __restrict__ b_wk,
    const float* __restrict__ W_wv, const float* __restrict__ b_wv,
    const float* __restrict__ W_out, const float* __restrict__ mem_init,
    __bf16* wih_bf, __bf16* whh_bf, __bf16* wk_bf, float* bk,
    __bf16* wout_bf, float* memf, __bf16* memb, float* invm,
    float* hf, __bf16* hb, __bf16* xcat, unsigned* bar) {
  const size_t tid = (size_t)blockIdx.x * blockDim.x + threadIdx.x;
  const size_t np  = (size_t)gridDim.x * blockDim.x;
  if (tid == 0) { bar[0] = 0u; bar[1] = 0u; }
  for (size_t i = tid; i < (size_t)kG3 * kX; i += np) wih_bf[i] = f2bf(W_ih[i]);
  for (size_t i = tid; i < (size_t)kG3 * kH; i += np) whh_bf[i] = f2bf(W_hh[i]);
  for (size_t i = tid; i < (size_t)kNK * kH; i += np) {
    size_t r = i / kH, c = i % kH;
    float v2 = (r < 256) ? W_rk[r * kH + c]
             : (r < 512) ? W_wk[(r - 256) * kH + c]
                         : W_wv[(r - 512) * kH + c];
    wk_bf[i] = f2bf(v2);
  }
  for (size_t i = tid; i < (size_t)kNK; i += np)
    bk[i] = (i < 256) ? b_rk[i] : (i < 512) ? b_wk[i - 256] : b_wv[i - 512];
  for (size_t i = tid; i < (size_t)kV * kH; i += np) wout_bf[i] = f2bf(W_out[i]);
  for (size_t i = tid; i < (size_t)kB * kS * kD; i += np) {
    float v2 = mem_init[i % ((size_t)kS * kD)];
    memf[i] = v2; memb[i] = f2bf(v2);
  }
  for (size_t i = tid; i < (size_t)kB * kS; i += np) {
    size_t s2 = i % kS;
    float p = 0.f;
    for (int d2 = 0; d2 < kD; ++d2) { float v2 = mem_init[s2 * kD + d2]; p += v2 * v2; }
    invm[i] = 1.f / (sqrtf(p) + kEPS);
  }
  for (size_t i = tid; i < (size_t)kB * kH; i += np) { hf[i] = 0.f; hb[i] = f2bf(0.f); }
  for (size_t i = tid; i < (size_t)kB * kX; i += np) {
    size_t b2 = i / kX, j = i % kX;
    float v2 = (j < kE) ? embed[(size_t)seq[b2 * kT] * kE + j] : 0.f;
    xcat[i] = f2bf(v2);
  }
}

// ---------------------------------------------------------------------------
// Main persistent kernel
// ---------------------------------------------------------------------------
__global__ __launch_bounds__(NTHR) void memnet_main(
    const int* __restrict__ seq, const float* __restrict__ embed,
    const float* __restrict__ b_ih, const float* __restrict__ b_hh,
    const float* __restrict__ W_er, const float* __restrict__ b_er,
    const float* __restrict__ W_ag, const float* __restrict__ b_ag,
    const float* __restrict__ b_out,
    const float* __restrict__ beta_read, const float* __restrict__ beta_write,
    const __bf16* __restrict__ wih_bf, const __bf16* __restrict__ whh_bf,
    const __bf16* __restrict__ wk_bf, const float* __restrict__ bk,
    const __bf16* __restrict__ wout_bf,
    float* memf, __bf16* memb, float* invm,
    float* gi, float* gh, float* hf, __bf16* hb, __bf16* xcat,
    unsigned* bar, float* out) {
  const int tid   = threadIdx.x;
  const int lane  = tid & 31, wv_ = tid >> 5;
  const bool hi   = lane >= 16;
  const int lrow  = lane & 15;
  const int wg    = blockIdx.x;
  const int gwave = wg * WPW + wv_;

  __shared__ __align__(32) float  ls_w[kNH][kS];
  __shared__ __align__(32) __bf16 ls_kn[16][kD];
  __shared__ float ls_keys[kNK];
  __shared__ float ls_erag[2 * kNH];
  __shared__ float ls_scale[kNH];
  __shared__ float ls_h[kH];
  __shared__ float ls_rv[kD];
  __shared__ int   ls_cnt[kNH], ls_cnt2[kNH];
  __shared__ int   ls_sl[kNH][CAP], ls_sl2[kNH][CAP];
  __shared__ float ls_wl[kNH][CAP], ls_wl2[kNH][CAP];
  __shared__ float ls_rs[kNH], ls_rs2[kNH];
  __shared__ float ls_keep[kS];
  __shared__ unsigned char ls_flag[kS];
  __shared__ int ls_union[kNH * CAP];
  __shared__ int ls_ucnt;

  for (int t = 0; t < kT; ++t) {
    // ---------- Region A: batched WMMA GEMMs (all WGs) ----------
    gemm32(xcat, kX, wih_bf, kX, b_ih, gi, kG3, kG3, kX, gwave);
    gemm32(hb,   kH, whh_bf, kH, b_hh, gh, kG3, kG3, kH, gwave);
    if (t > 0)
      gemm32(hb, kH, wout_bf, kH, b_out, out + (size_t)(t - 1) * kV,
             (size_t)kT * kV, kV, kH, gwave);
    gsync(bar);
    // ---------- Region B: per-batch recurrence (WG b < 32) ----------
    if (wg < kB) {
      const int b2 = wg;
      const float* giB = gi + (size_t)b2 * kG3;
      const float* ghB = gh + (size_t)b2 * kG3;
      float*  hfB = hf + (size_t)b2 * kH;
      __bf16* hbB = hb + (size_t)b2 * kH;
      // GRU gates
      for (int j = tid; j < kH; j += NTHR) {
        float r  = sigmoidf(giB[j] + ghB[j]);
        float z  = sigmoidf(giB[kH + j] + ghB[kH + j]);
        float nn = tanhf(giB[2 * kH + j] + r * ghB[2 * kH + j]);
        float hv = (1.f - z) * nn + z * hfB[j];
        hfB[j] = hv; hbB[j] = f2bf(hv); ls_h[j] = hv;
      }
      __syncthreads();
      // keys rk|wk|wv : WMMA GEMV (A rows = replicated h)
      {
        const __bf16* ar = hbB + (hi ? 16 : 0);
        for (int nt = wv_; nt < kNK / 16; nt += WPW) {
          const __bf16* wr = wk_bf + (size_t)(nt * 16 + lrow) * kH + (hi ? 16 : 0);
          v8f acc = {};
          for (int kc = 0; kc < kH; kc += 32) {
            ABf a, b;
            a.v    = *(const v16bf*)(ar + kc);
            b.v    = *(const v16bf*)(wr + kc);
            acc = __builtin_amdgcn_wmma_f32_16x16x32_bf16(false, a.v, false, b.v,
                                                          (short)0, acc, false, false);
          }
          if (lane < 16) { int nx = nt * 16 + lane; ls_keys[nx] = acc[0] + bk[nx]; }
        }
      }
      __syncthreads();
      // er/ag gates: one wave per output, shfl reduce
      {
        const float* wrow2 = (wv_ < kNH) ? (W_er + (size_t)wv_ * kH)
                                         : (W_ag + (size_t)(wv_ - kNH) * kH);
        float p = 0.f;
        for (int k2 = lane; k2 < kH; k2 += 32) p += ls_h[k2] * wrow2[k2];
        p = wredSum(p);
        if (lane == 0) {
          float bb = (wv_ < kNH) ? b_er[wv_] : b_ag[wv_ - kNH];
          ls_erag[wv_] = sigmoidf(p + bb);
        }
      }
      __syncthreads();
      const __bf16* memBb = memb + (size_t)b2 * kS * kD;
      float* memFb  = memf + (size_t)b2 * kS * kD;
      float* invmBp = invm + (size_t)b2 * kS;
      // read addressing (pre-write memory)
      do_address(ls_keys, beta_read, memBb, invmBp, ls_w, ls_kn, ls_scale,
                 ls_cnt, ls_sl, ls_wl, ls_rs);
      // read vector (sparse gather over kept slots)
      if (tid < kD) {
        float acc = 0.f;
        for (int h2 = 0; h2 < kNH; ++h2) {
          float inv = 1.f / (ls_rs[h2] + kEPS);
          int c = ls_cnt[h2]; if (c > CAP) c = CAP;
          for (int i = 0; i < c; ++i)
            acc += ls_wl[h2][i] * inv * memFb[(size_t)ls_sl[h2][i] * kD + tid];
        }
        ls_rv[tid] = acc * (1.f / kNH);
      }
      // write addressing (also pre-write memory)
      do_address(ls_keys + kNH * kD, beta_write, memBb, invmBp, ls_w, ls_kn,
                 ls_scale, ls_cnt2, ls_sl2, ls_wl2, ls_rs2);
      // ---- sparse erase/add memory update ----
      if (tid == 0) ls_ucnt = 0;
      for (int s2 = tid; s2 < kS; s2 += NTHR) ls_flag[s2] = 0;
      __syncthreads();
      for (int h2 = 0; h2 < kNH; ++h2) {       // keep[s] = prod_h (1 - w'*er)
        float er  = ls_erag[h2];
        float inv = 1.f / (ls_rs2[h2] + kEPS);
        int c = ls_cnt2[h2]; if (c > CAP) c = CAP;
        for (int i = tid; i < c; i += NTHR) {
          int s2 = ls_sl2[h2][i];
          float f = 1.f - ls_wl2[h2][i] * inv * er;
          if (!ls_flag[s2]) {
            ls_flag[s2] = 1;
            int u = atomicAdd(&ls_ucnt, 1);
            ls_union[u] = s2;
            ls_keep[s2] = f;
          } else ls_keep[s2] *= f;
        }
        __syncthreads();
      }
      const int uc = ls_ucnt;
      for (int i = tid; i < uc * kD; i += NTHR) {   // erase
        int u = i >> 6, d2 = i & 63; int s2 = ls_union[u];
        memFb[(size_t)s2 * kD + d2] *= ls_keep[s2];
      }
      __syncthreads();
      for (int h2 = 0; h2 < kNH; ++h2) {            // add
        float ag  = ls_erag[kNH + h2];
        float inv = 1.f / (ls_rs2[h2] + kEPS);
        int c = ls_cnt2[h2]; if (c > CAP) c = CAP;
        for (int i = tid; i < c * kD; i += NTHR) {
          int ii = i >> 6, d2 = i & 63; int s2 = ls_sl2[h2][ii];
          memFb[(size_t)s2 * kD + d2] +=
              ls_wl2[h2][ii] * inv * ag * ls_keys[2 * kNH * kD + h2 * kD + d2];
        }
        __syncthreads();
      }
      // incremental norm + bf16 mirror upkeep: one wave per touched row
      for (int u = wv_; u < uc; u += WPW) {
        int s2 = ls_union[u];
        float m0 = memFb[(size_t)s2 * kD + 2 * lane];
        float m1 = memFb[(size_t)s2 * kD + 2 * lane + 1];
        float p = wredSum(m0 * m0 + m1 * m1);
        if (lane == 0) invmBp[s2] = 1.f / (sqrtf(p) + kEPS);
        memb[(size_t)b2 * kS * kD + (size_t)s2 * kD + 2 * lane]     = f2bf(m0);
        memb[(size_t)b2 * kS * kD + (size_t)s2 * kD + 2 * lane + 1] = f2bf(m1);
      }
      // build GRU input for next step: [emb(x_{t+1}) | rv_t] in bf16
      __bf16* xcB = xcat + (size_t)b2 * kX;
      if (t + 1 < kT) {
        int tok = seq[b2 * kT + t + 1];
        const float* er2 = embed + (size_t)tok * kE;
        for (int j = tid; j < kE; j += NTHR) xcB[j] = f2bf(er2[j]);
      }
      for (int j = tid; j < kD; j += NTHR) xcB[kE + j] = f2bf(ls_rv[j]);
    }
    gsync(bar);
  }
  // epilogue: logits for final timestep
  gemm32(hb, kH, wout_bf, kH, b_out, out + (size_t)(kT - 1) * kV,
         (size_t)kT * kV, kV, kH, gwave);
}

// ---------------------------------------------------------------------------
extern "C" void kernel_launch(void* const* d_in, const int* in_sizes, int n_in,
                              void* d_out, int out_size, void* d_ws, size_t ws_size,
                              hipStream_t stream) {
  const int*   seq       = (const int*)d_in[0];
  const float* embed     = (const float*)d_in[1];
  const float* W_ih      = (const float*)d_in[2];
  const float* W_hh      = (const float*)d_in[3];
  const float* b_ih      = (const float*)d_in[4];
  const float* b_hh      = (const float*)d_in[5];
  const float* W_rk      = (const float*)d_in[6];
  const float* b_rk      = (const float*)d_in[7];
  const float* W_wk      = (const float*)d_in[8];
  const float* b_wk      = (const float*)d_in[9];
  const float* W_wv      = (const float*)d_in[10];
  const float* b_wv      = (const float*)d_in[11];
  const float* W_er      = (const float*)d_in[12];
  const float* b_er      = (const float*)d_in[13];
  const float* W_ag      = (const float*)d_in[14];
  const float* b_ag      = (const float*)d_in[15];
  const float* W_out     = (const float*)d_in[16];
  const float* b_out     = (const float*)d_in[17];
  const float* beta_read = (const float*)d_in[18];
  const float* beta_write= (const float*)d_in[19];
  const float* mem_init  = (const float*)d_in[20];
  (void)in_sizes; (void)n_in; (void)out_size;   // topk (d_in[21]) hardcoded = 8

  char* ws = (char*)d_ws;
  size_t off = 0;
  auto alloc = [&](size_t bytes) -> void* {
    void* p = ws + off;
    off = (off + bytes + 255) & ~(size_t)255;
    return p;
  };
  unsigned* bar   = (unsigned*)alloc(256);
  __bf16* wih_bf  = (__bf16*)alloc((size_t)kG3 * kX * 2);
  __bf16* whh_bf  = (__bf16*)alloc((size_t)kG3 * kH * 2);
  __bf16* wk_bf   = (__bf16*)alloc((size_t)kNK * kH * 2);
  float*  bk      = (float*) alloc((size_t)kNK * 4);
  __bf16* wout_bf = (__bf16*)alloc((size_t)kV * kH * 2);
  float*  memf    = (float*) alloc((size_t)kB * kS * kD * 4);
  __bf16* memb    = (__bf16*)alloc((size_t)kB * kS * kD * 2);
  float*  invm    = (float*) alloc((size_t)kB * kS * 4);
  float*  gi      = (float*) alloc((size_t)kB * kG3 * 4);
  float*  gh      = (float*) alloc((size_t)kB * kG3 * 4);
  float*  hf      = (float*) alloc((size_t)kB * kH * 4);
  __bf16* hb      = (__bf16*)alloc((size_t)kB * kH * 2);
  __bf16* xcat    = (__bf16*)alloc((size_t)kB * kX * 2);
  if (off > ws_size) return;  // workspace too small (~26 MB needed)

  prep_kernel<<<1024, 256, 0, stream>>>(
      seq, embed, W_ih, W_hh, W_rk, b_rk, W_wk, b_wk, W_wv, b_wv, W_out,
      mem_init, wih_bf, whh_bf, wk_bf, bk, wout_bf, memf, memb, invm,
      hf, hb, xcat, bar);
  memnet_main<<<NWG, NTHR, 0, stream>>>(
      seq, embed, b_ih, b_hh, W_er, b_er, W_ag, b_ag, b_out,
      beta_read, beta_write, wih_bf, whh_bf, wk_bf, bk, wout_bf,
      memf, memb, invm, gi, gh, hf, hb, xcat, bar, (float*)d_out);
}
